// Attention_72404558676587
// MI455X (gfx1250) — compile-verified, matrix operand order checked by reading, not run on hardware
//
#include <hip/hip_runtime.h>
#include <hip/hip_bf16.h>
#include <math.h>

// Problem constants (B=1)
#define T_SEQ  2048
#define DMODEL 2048
#define NHEADS 16
#define NKV    8
#define HDIM   128

typedef __attribute__((ext_vector_type(16))) _Float16 v16h;
typedef __attribute__((ext_vector_type(8)))  _Float16 v8h;
typedef __attribute__((ext_vector_type(8)))  float    v8f;

// ---------------------------------------------------------------------------
// Fragment loader: A-layout 16x32 f16 fragment from a row-major matrix.
// Lane L<16 holds row (row0+L), K in {k0..k0+7, k0+16..k0+23};
// Lane L>=16 holds row (row0+L-16), K in {k0+8..k0+15, k0+24..k0+31}.
// Two 16-byte loads per lane. Also used for B fragments by loading B^T rows,
// and works on LDS pointers (clang infers addrspace(3) -> ds_load_b128).
// ---------------------------------------------------------------------------
static __device__ __forceinline__ v16h load_frag16(const _Float16* base, int ld,
                                                   int row0, int k0) {
  const int lane = threadIdx.x & 31;
  const int r  = row0 + (lane & 15);
  const int kg = k0 + ((lane >> 4) << 3);
  const _Float16* p = base + (size_t)r * (size_t)ld + kg;
  v8h lo = *(const v8h*)(p);
  v8h hi = *(const v8h*)(p + 16);
  v16h out;
#pragma unroll
  for (int i = 0; i < 8; ++i) { out[i] = lo[i]; out[i + 8] = hi[i]; }
  return out;
}

// Store a 16x16 f32 accumulator tile (C layout) with bias add.
static __device__ __forceinline__ void store_tile(float* C, int ldc, int row0,
                                                  int col0, v8f c,
                                                  const float* bias) {
  const int lane = threadIdx.x & 31;
  const int n = col0 + (lane & 15);
  const int m = row0 + ((lane >> 4) << 3);
  const float b = bias ? bias[n] : 0.0f;
#pragma unroll
  for (int v = 0; v < 8; ++v) C[(size_t)(m + v) * ldc + n] = c[v] + b;
}

// ---------------------------------------------------------------------------
// f32 -> f16 convert / transpose kernels
// ---------------------------------------------------------------------------
__global__ void f32_to_f16_kernel(const float* __restrict__ in,
                                  _Float16* __restrict__ out, int n) {
  int i = blockIdx.x * blockDim.x + threadIdx.x;
  if (i < n) out[i] = (_Float16)in[i];
}

// W: K x N row-major (f32) -> Wt: N x K row-major (f16)
__global__ void transpose_f32_to_f16(const float* __restrict__ W,
                                     _Float16* __restrict__ Wt, int K, int N) {
  int i = blockIdx.x * blockDim.x + threadIdx.x;
  if (i < K * N) {
    int n = i / K;
    int k = i - n * K;
    Wt[(size_t)n * K + k] = (_Float16)W[(size_t)k * N + n];
  }
}

// ---------------------------------------------------------------------------
// WMMA GEMM with LDS staging:
//   C(MxN,f32) = A(MxK,f16 row-major) * Bt(NxK,f16 row-major)^T + bias
// block = 256 threads (8 waves), block tile 128x128, K-step 32.
// Each K-step stages 128x32 A and Bt tiles into LDS (padded stride keeps
// 16B alignment), then each wave computes a 32x64 tile: 8 WMMAs from
// 6 LDS fragment reads -> high reuse, low L2 pressure.
// ---------------------------------------------------------------------------
#define BM 128
#define BN 128
#define BK 32
#define LDSTR 40  // 32 + 8 pad f16; row stride 80B (multiple of 16B)

__global__ __launch_bounds__(256) void gemm_f16_wmma(
    const _Float16* __restrict__ A, const _Float16* __restrict__ Bt,
    const float* __restrict__ bias, float* __restrict__ C, int M, int N,
    int K) {
  __shared__ _Float16 Ash[BM * LDSTR];
  __shared__ _Float16 Bsh[BN * LDSTR];

  const int tid  = threadIdx.x;
  const int wave = tid >> 5;
  const int rowblk = blockIdx.y * BM;
  const int colblk = blockIdx.x * BN;
  const int wr = (wave >> 1) * 32;  // wave row offset within block tile
  const int wc = (wave & 1) * 64;   // wave col offset within block tile

  v8f zero = {};
  v8f acc[2][4];
#pragma unroll
  for (int i = 0; i < 2; ++i)
#pragma unroll
    for (int j = 0; j < 4; ++j) acc[i][j] = zero;

  for (int k0 = 0; k0 < K; k0 += BK) {
    __syncthreads();  // WAR: previous iteration's fragment reads done
    // Stage tiles: 128 rows x 32 k each = 512 x v8h per tile, 2 per thread.
#pragma unroll
    for (int it = 0; it < 2; ++it) {
      const int idx = tid + it * 256;
      const int r = idx >> 2;
      const int kk = (idx & 3) * 8;
      *(v8h*)(&Ash[r * LDSTR + kk]) =
          *(const v8h*)(A + (size_t)(rowblk + r) * K + k0 + kk);
      *(v8h*)(&Bsh[r * LDSTR + kk]) =
          *(const v8h*)(Bt + (size_t)(colblk + r) * K + k0 + kk);
    }
    __syncthreads();

    v16h a0 = load_frag16(Ash, LDSTR, wr, 0);
    v16h a1 = load_frag16(Ash, LDSTR, wr + 16, 0);
    v16h b0 = load_frag16(Bsh, LDSTR, wc, 0);
    v16h b1 = load_frag16(Bsh, LDSTR, wc + 16, 0);
    v16h b2 = load_frag16(Bsh, LDSTR, wc + 32, 0);
    v16h b3 = load_frag16(Bsh, LDSTR, wc + 48, 0);

    acc[0][0] = __builtin_amdgcn_wmma_f32_16x16x32_f16(false, a0, false, b0,
                                                       (short)0, acc[0][0],
                                                       false, false);
    acc[0][1] = __builtin_amdgcn_wmma_f32_16x16x32_f16(false, a0, false, b1,
                                                       (short)0, acc[0][1],
                                                       false, false);
    acc[0][2] = __builtin_amdgcn_wmma_f32_16x16x32_f16(false, a0, false, b2,
                                                       (short)0, acc[0][2],
                                                       false, false);
    acc[0][3] = __builtin_amdgcn_wmma_f32_16x16x32_f16(false, a0, false, b3,
                                                       (short)0, acc[0][3],
                                                       false, false);
    acc[1][0] = __builtin_amdgcn_wmma_f32_16x16x32_f16(false, a1, false, b0,
                                                       (short)0, acc[1][0],
                                                       false, false);
    acc[1][1] = __builtin_amdgcn_wmma_f32_16x16x32_f16(false, a1, false, b1,
                                                       (short)0, acc[1][1],
                                                       false, false);
    acc[1][2] = __builtin_amdgcn_wmma_f32_16x16x32_f16(false, a1, false, b2,
                                                       (short)0, acc[1][2],
                                                       false, false);
    acc[1][3] = __builtin_amdgcn_wmma_f32_16x16x32_f16(false, a1, false, b3,
                                                       (short)0, acc[1][3],
                                                       false, false);
  }

#pragma unroll
  for (int i = 0; i < 2; ++i)
#pragma unroll
    for (int j = 0; j < 4; ++j)
      store_tile(C, N, rowblk + wr + i * 16, colblk + wc + j * 16, acc[i][j],
                 bias);
}

// ---------------------------------------------------------------------------
// RoPE + pack: Qf32(T x 2048), KVf32(T x 2048) ->
//   Qh  f16 [NHEADS][T][HDIM]   (rotary applied, pre-scaled by 1/sqrt(HDIM))
//   Kh  f16 [NKV][T][HDIM]      (rotary applied)
//   Vt  f16 [NKV][HDIM][T]      (transposed: d-major, for contiguous B frags)
// ---------------------------------------------------------------------------
__global__ __launch_bounds__(256) void rope_pack_kernel(
    const float* __restrict__ Q, const float* __restrict__ KV,
    _Float16* __restrict__ Qh, _Float16* __restrict__ Kh,
    _Float16* __restrict__ Vt) {
  const int t = blockIdx.x;
  const float qscale = 0.08838834764831845f;  // 1/sqrt(128)

  for (int i = threadIdx.x; i < NHEADS * (HDIM / 2); i += blockDim.x) {
    int h = i >> 6, d = i & 63;
    float x1 = Q[(size_t)t * DMODEL + h * HDIM + 2 * d];
    float x2 = Q[(size_t)t * DMODEL + h * HDIM + 2 * d + 1];
    float inv = __powf(10000.0f, -(float)(2 * d) * (1.0f / HDIM));
    float s, c;
    __sincosf((float)t * inv, &s, &c);
    _Float16* q = Qh + ((size_t)h * T_SEQ + t) * HDIM;
    q[2 * d]     = (_Float16)((x1 * c - x2 * s) * qscale);
    q[2 * d + 1] = (_Float16)((x1 * s + x2 * c) * qscale);
  }
  for (int i = threadIdx.x; i < NKV * (HDIM / 2); i += blockDim.x) {
    int h = i >> 6, d = i & 63;
    float x1 = KV[(size_t)t * DMODEL + h * HDIM + 2 * d];
    float x2 = KV[(size_t)t * DMODEL + h * HDIM + 2 * d + 1];
    float inv = __powf(10000.0f, -(float)(2 * d) * (1.0f / HDIM));
    float s, c;
    __sincosf((float)t * inv, &s, &c);
    _Float16* k = Kh + ((size_t)h * T_SEQ + t) * HDIM;
    k[2 * d]     = (_Float16)(x1 * c - x2 * s);
    k[2 * d + 1] = (_Float16)(x1 * s + x2 * c);
  }
  for (int i = threadIdx.x; i < NKV * HDIM; i += blockDim.x) {
    int h = i >> 7, d = i & 127;
    Vt[((size_t)h * HDIM + d) * T_SEQ + t] =
        (_Float16)KV[(size_t)t * DMODEL + (NKV + h) * HDIM + d];
  }
}

// ---------------------------------------------------------------------------
// Flash attention (causal, GQA head h uses KV head h/2).
// grid = (T/64, NHEADS); block = 128 (4 waves); wave owns 16 query rows.
// Scores via WMMA, online softmax with 16-lane shuffle reductions,
// P routed through per-wave LDS tile to convert C-layout -> A-layout.
// Output: f16 [T][DMODEL] (t-major), already merged across heads.
// ---------------------------------------------------------------------------
__global__ __launch_bounds__(128) void flash_attn_kernel(
    const _Float16* __restrict__ Qh, const _Float16* __restrict__ Kh,
    const _Float16* __restrict__ Vt, _Float16* __restrict__ O) {
  __shared__ _Float16 pbuf_all[4 * 16 * 32];
  const int wave = threadIdx.x >> 5;
  const int lane = threadIdx.x & 31;
  const int h = blockIdx.y;
  const int q0 = blockIdx.x * 64 + wave * 16;

  const _Float16* Qp = Qh + (size_t)h * T_SEQ * HDIM;
  const _Float16* Kp = Kh + (size_t)(h >> 1) * T_SEQ * HDIM;
  const _Float16* Vp = Vt + (size_t)(h >> 1) * HDIM * T_SEQ;
  _Float16* pb = pbuf_all + wave * (16 * 32);

  v16h qf[4];
#pragma unroll
  for (int kk = 0; kk < 4; ++kk) qf[kk] = load_frag16(Qp, HDIM, q0, kk * 32);

  v8f o[8];
  v8f zero = {};
#pragma unroll
  for (int i = 0; i < 8; ++i) o[i] = zero;
  float mrow[8], lrow[8];
#pragma unroll
  for (int v = 0; v < 8; ++v) { mrow[v] = -3.0e38f; lrow[v] = 0.0f; }

  const int n  = lane & 15;
  const int mb = (lane >> 4) << 3;

  const int kmax = q0 + 15;  // last valid key for this wave's rows
  for (int kb = 0; kb <= kmax; kb += 32) {
    // ---- scores: 16 q-rows x 32 keys ----
    v8f s0 = zero, s1 = zero;
#pragma unroll
    for (int kk = 0; kk < 4; ++kk) {
      v16h k0 = load_frag16(Kp, HDIM, kb, kk * 32);
      v16h k1 = load_frag16(Kp, HDIM, kb + 16, kk * 32);
      s0 = __builtin_amdgcn_wmma_f32_16x16x32_f16(false, qf[kk], false, k0,
                                                  (short)0, s0, false, false);
      s1 = __builtin_amdgcn_wmma_f32_16x16x32_f16(false, qf[kk], false, k1,
                                                  (short)0, s1, false, false);
    }
    // ---- causal mask + online softmax ----
#pragma unroll
    for (int v = 0; v < 8; ++v) {
      const int row = q0 + mb + v;
      float a = (kb + n      <= row) ? s0[v] : -3.0e38f;
      float b = (kb + 16 + n <= row) ? s1[v] : -3.0e38f;
      float m = fmaxf(a, b);
#pragma unroll
      for (int off = 1; off < 16; off <<= 1) m = fmaxf(m, __shfl_xor(m, off));
      float mnew  = fmaxf(mrow[v], m);
      float scale = __expf(mrow[v] - mnew);
      float p0 = __expf(a - mnew);
      float p1 = __expf(b - mnew);
      float psum = p0 + p1;
#pragma unroll
      for (int off = 1; off < 16; off <<= 1) psum += __shfl_xor(psum, off);
      lrow[v] = lrow[v] * scale + psum;
      mrow[v] = mnew;
#pragma unroll
      for (int tdt = 0; tdt < 8; ++tdt) o[tdt][v] *= scale;
      // stage P (C-layout position) into this wave's LDS tile
      pb[(mb + v) * 32 + n]      = (_Float16)p0;
      pb[(mb + v) * 32 + n + 16] = (_Float16)p1;
    }
    asm volatile("s_wait_dscnt 0x0" ::: "memory");  // wave-local LDS fence
    v16h pa = load_frag16(pb, 32, 0, 0);  // re-read P in A-fragment layout
    // ---- O += P(16x32) @ V(32x128) ----
#pragma unroll
    for (int tdt = 0; tdt < 8; ++tdt) {
      v16h vb = load_frag16(Vp, T_SEQ, tdt * 16, kb);
      o[tdt] = __builtin_amdgcn_wmma_f32_16x16x32_f16(false, pa, false, vb,
                                                      (short)0, o[tdt], false,
                                                      false);
    }
  }
  // ---- normalize + write merged f16 output [T][DMODEL] ----
#pragma unroll
  for (int v = 0; v < 8; ++v) {
    const float inv = 1.0f / lrow[v];
    const int row = q0 + mb + v;
#pragma unroll
    for (int tdt = 0; tdt < 8; ++tdt) {
      O[(size_t)row * DMODEL + h * HDIM + tdt * 16 + n] =
          (_Float16)(o[tdt][v] * inv);
    }
  }
}

// ---------------------------------------------------------------------------
// Host-side orchestration
// ---------------------------------------------------------------------------
extern "C" void kernel_launch(void* const* d_in, const int* in_sizes, int n_in,
                              void* d_out, int out_size, void* d_ws,
                              size_t ws_size, hipStream_t stream) {
  (void)in_sizes; (void)n_in; (void)out_size; (void)ws_size;
  const float* x   = (const float*)d_in[0];
  const float* Wq  = (const float*)d_in[1];
  const float* bq  = (const float*)d_in[2];
  const float* Wkv = (const float*)d_in[3];
  const float* bkv = (const float*)d_in[4];
  const float* Wo  = (const float*)d_in[5];
  const float* bo  = (const float*)d_in[6];
  float* out = (float*)d_out;

  char* ws = (char*)d_ws;
  size_t off = 0;
  auto alloc = [&](size_t bytes) -> char* {
    char* p = ws + off;
    off += (bytes + 255) & ~(size_t)255;
    return p;
  };
  const size_t TD = (size_t)T_SEQ * DMODEL;
  _Float16* xh    = (_Float16*)alloc(TD * 2);                     // x in f16
  _Float16* WqT   = (_Float16*)alloc(TD * 2);                     // Wq^T f16
  _Float16* WkvT  = (_Float16*)alloc(TD * 2);                     // Wkv^T f16
  _Float16* WoT   = (_Float16*)alloc(TD * 2);                     // Wo^T f16
  float*    Qf    = (float*)alloc(TD * 4);                        // q fp32
  float*    KVf   = (float*)alloc(TD * 4);                        // kv fp32
  _Float16* Qh    = (_Float16*)alloc((size_t)NHEADS * T_SEQ * HDIM * 2);
  _Float16* Khp   = (_Float16*)alloc((size_t)NKV * T_SEQ * HDIM * 2);
  _Float16* Vt    = (_Float16*)alloc((size_t)NKV * HDIM * T_SEQ * 2);
  _Float16* AttnO = (_Float16*)alloc(TD * 2);                     // attn out f16

  const int nel = (int)TD;
  f32_to_f16_kernel<<<(nel + 255) / 256, 256, 0, stream>>>(x, xh, nel);
  transpose_f32_to_f16<<<(nel + 255) / 256, 256, 0, stream>>>(Wq, WqT, DMODEL,
                                                              DMODEL);
  transpose_f32_to_f16<<<(nel + 255) / 256, 256, 0, stream>>>(Wkv, WkvT,
                                                              DMODEL, DMODEL);
  transpose_f32_to_f16<<<(nel + 255) / 256, 256, 0, stream>>>(Wo, WoT, DMODEL,
                                                              DMODEL);

  dim3 ggrid(DMODEL / BN, T_SEQ / BM);
  gemm_f16_wmma<<<ggrid, 256, 0, stream>>>(xh, WqT, bq, Qf, T_SEQ, DMODEL,
                                           DMODEL);
  gemm_f16_wmma<<<ggrid, 256, 0, stream>>>(xh, WkvT, bkv, KVf, T_SEQ, DMODEL,
                                           DMODEL);

  rope_pack_kernel<<<T_SEQ, 256, 0, stream>>>(Qf, KVf, Qh, Khp, Vt);

  flash_attn_kernel<<<dim3(T_SEQ / 64, NHEADS), 128, 0, stream>>>(Qh, Khp, Vt,
                                                                  AttnO);

  gemm_f16_wmma<<<ggrid, 256, 0, stream>>>(AttnO, WoT, bo, out, T_SEQ, DMODEL,
                                           DMODEL);
}
